// D_R_43748536877439
// MI455X (gfx1250) — matrix operand order repaired and checked by reading, not compiled, not run on hardware
//
#include <hip/hip_runtime.h>
#include <math.h>

// ---------------------------------------------------------------------------
// D_R hierarchical decomposition model on MI455X (gfx1250).
// All activations/weights live in HBM as bf16; every contraction runs through
// v_wmma_f32_16x16x32_bf16 with f32 accumulation. LDS tiles are filled with
// async global->LDS b128 copies (ASYNCcnt) when the toolchain exposes them.
// ---------------------------------------------------------------------------

#define B_   64
#define N_   64
#define T_   512
#define H_   1024
#define K_   4
#define NBR_ 16
#define EPS_ 1e-5f

typedef unsigned short bfu;   // raw bf16 storage
typedef __attribute__((ext_vector_type(16))) __bf16 v16bf;
typedef __attribute__((ext_vector_type(8)))  float  v8f;
typedef __attribute__((ext_vector_type(4)))  int    v4i;

#define AS1 __attribute__((address_space(1)))
#define AS3 __attribute__((address_space(3)))

#if defined(__has_builtin)
#if __has_builtin(__builtin_amdgcn_global_load_async_to_lds_b128)
#define HAVE_ASYNC_LDS 1
#endif
#endif

__device__ __forceinline__ float b2f(bfu h) {
  union { float f; unsigned u; } v; v.u = (unsigned)h << 16; return v.f;
}
__device__ __forceinline__ bfu f2b(float f) {
  union { float f; unsigned u; } v; v.f = f;
  unsigned x = v.u;
  x = (x + 0x7FFFu + ((x >> 16) & 1u)) >> 16;   // RNE
  return (bfu)x;
}
__device__ __forceinline__ float leaky_(float v) { return v >= 0.f ? v : 0.01f * v; }

__device__ __forceinline__ unsigned hash_u32(unsigned x) {
  x ^= x >> 16; x *= 0x7feb352dU; x ^= x >> 15; x *= 0x846ca68bU; x ^= x >> 16;
  return x;
}

// fp32 -> bf16 bulk conversion (weights / input, once per launch)
__global__ void k_f2b(const float* __restrict__ src, bfu* __restrict__ dst, long n)
{
  long i = (long)blockIdx.x * blockDim.x + threadIdx.x;
  if (i < n) dst[i] = f2b(src[i]);
}

// ---------------------------------------------------------------------------
// WMMA GEMM (bf16 in, f32 accum):  C[m,n] = sum_k A[m,k]*W[n,k] (+bias[n])
// (+resid[m,n]) [leaky].  Block tile 128x64x64, 8 wave32 waves, each wave a
// 32x32 tile as 2x2 WMMA fragments; 2 K-substeps (8 v_wmma) per LDS refill.
// ---------------------------------------------------------------------------
#define BM 128
#define BN 64
#define BK 64
#define SCOL (BK / 2 + 4)   // 36 dwords/row = 144B: 16B-aligned, conflict-free

__global__ __launch_bounds__(256)
void k_gemm(const bfu* __restrict__ A, long lda,
            const bfu* __restrict__ W, long ldw,
            const float* __restrict__ bias,
            const bfu* __restrict__ resid, long ldr,
            void* __restrict__ Cout, long ldc,
            int K, int epi, int outBf16)
{
  __shared__ unsigned sA[BM][SCOL];
  __shared__ unsigned sB[BN][SCOL];

  const int tid  = threadIdx.x;
  const int lane = tid & 31;
  const int wave = tid >> 5;
  const int wm   = wave >> 1;           // 0..3  (M)
  const int wn   = wave & 1;            // 0..1  (N)
  const long rowBase = (long)blockIdx.x * BM;
  const long colBase = (long)blockIdx.y * BN;

  v8f acc[2][2];
#pragma unroll
  for (int i = 0; i < 2; ++i)
#pragma unroll
    for (int j = 0; j < 2; ++j)
#pragma unroll
      for (int r = 0; r < 8; ++r) acc[i][j][r] = 0.f;

  const int half = lane >> 4;
  const int l15  = lane & 15;

  for (int kk = 0; kk < K; kk += BK) {
    const bfu* Ab = A + rowBase * lda + kk;
    const bfu* Wb = W + colBase * ldw + kk;

#ifdef HAVE_ASYNC_LDS
    // async DMA global->LDS, 16B chunks, tracked by ASYNCcnt
#pragma unroll
    for (int p = tid; p < BM * (BK / 8); p += 256) {       // 1024 chunks
      int row = p >> 3, c4 = p & 7;
      __builtin_amdgcn_global_load_async_to_lds_b128(
          (AS1 v4i*)(Ab + (long)row * lda + 8 * c4),
          (AS3 v4i*)&sA[row][4 * c4], 0, 0);
    }
#pragma unroll
    for (int p = tid; p < BN * (BK / 8); p += 256) {       // 512 chunks
      int row = p >> 3, c4 = p & 7;
      __builtin_amdgcn_global_load_async_to_lds_b128(
          (AS1 v4i*)(Wb + (long)row * ldw + 8 * c4),
          (AS3 v4i*)&sB[row][4 * c4], 0, 0);
    }
#if __has_builtin(__builtin_amdgcn_s_wait_asynccnt)
    __builtin_amdgcn_s_wait_asynccnt(0);
#else
    asm volatile("s_wait_asynccnt 0" ::: "memory");
#endif
#else
    // fallback: b128 load + ds_store_b128 (data already bf16 -> no VALU pack)
#pragma unroll
    for (int p = tid; p < BM * (BK / 8); p += 256) {
      int row = p >> 3, c4 = p & 7;
      *(uint4*)&sA[row][4 * c4] = *(const uint4*)(Ab + (long)row * lda + 8 * c4);
    }
#pragma unroll
    for (int p = tid; p < BN * (BK / 8); p += 256) {
      int row = p >> 3, c4 = p & 7;
      *(uint4*)&sB[row][4 * c4] = *(const uint4*)(Wb + (long)row * ldw + 8 * c4);
    }
#endif
    if (kk + BK < K)  // global_prefetch_b8 of next A tile
      __builtin_prefetch(Ab + BK + (long)(tid & (BM - 1)) * lda, 0, 1);
    __syncthreads();

#pragma unroll
    for (int s = 0; s < 2; ++s) {       // two 16x16x32 K-substeps
      const int c0 = s << 4;            // 16 dword columns per substep
      union { v16bf v; unsigned u[8]; } a0, a1, b0, b1;
#pragma unroll
      for (int v = 0; v < 8; ++v) {
        // A 16x32 fragment layout (ISA 7.12.2)
        int ca = c0 + ((v & 4) << 1) + (v & 3) + (half << 2);
        a0.u[v] = sA[wm * 32 + l15][ca];
        a1.u[v] = sA[wm * 32 + 16 + l15][ca];
        // B 32x16 layout: same K order as A fragment
        int cb = c0 + ((v & 4) << 1) + (v & 3) + (half << 2);
        b0.u[v] = sB[wn * 32 + l15][cb];
        b1.u[v] = sB[wn * 32 + 16 + l15][cb];
      }
      acc[0][0] = __builtin_amdgcn_wmma_f32_16x16x32_bf16(false, a0.v, false, b0.v,
                                                          (short)0, acc[0][0], false, false);
      acc[0][1] = __builtin_amdgcn_wmma_f32_16x16x32_bf16(false, a0.v, false, b1.v,
                                                          (short)0, acc[0][1], false, false);
      acc[1][0] = __builtin_amdgcn_wmma_f32_16x16x32_bf16(false, a1.v, false, b0.v,
                                                          (short)0, acc[1][0], false, false);
      acc[1][1] = __builtin_amdgcn_wmma_f32_16x16x32_bf16(false, a1.v, false, b1.v,
                                                          (short)0, acc[1][1], false, false);
    }
    __syncthreads();
  }

  // C/D layout: VGPR r -> M = r + 8*(lane>=16), N = lane&15
  const int cl = lane & 15, ch = lane >> 4;
#pragma unroll
  for (int mi = 0; mi < 2; ++mi)
#pragma unroll
    for (int ni = 0; ni < 2; ++ni)
#pragma unroll
      for (int r = 0; r < 8; ++r) {
        long m = rowBase + wm * 32 + mi * 16 + ch * 8 + r;
        long n = colBase + wn * 32 + ni * 16 + cl;
        float v = acc[mi][ni][r];
        if (bias)  v += bias[n];
        if (resid) v += b2f(resid[m * ldr + n]);
        if (epi)   v = leaky_(v);
        if (outBf16) ((bfu*)Cout)[m * ldc + n] = f2b(v);
        else         ((float*)Cout)[m * ldc + n] = v;
      }
}

// ---------------------------------------------------------------------------
// mcd: 3-tap conv mask (K channels) then x*mask; strided out for total slices
// ---------------------------------------------------------------------------
__global__ void k_mcd(const bfu* __restrict__ x, const float* __restrict__ w,
                      const float* __restrict__ b, bfu* __restrict__ out,
                      long outRowStride)
{
  long idx = (long)blockIdx.x * blockDim.x + threadIdx.x;
  const long total = (long)B_ * N_ * T_;
  if (idx >= total) return;
  long bn = idx / T_; int t = (int)(idx % T_);
  const bfu* xr = x + bn * T_;
  float xm = t > 0      ? b2f(xr[t - 1]) : 0.f;
  float x0 =              b2f(xr[t]);
  float xp = t < T_ - 1 ? b2f(xr[t + 1]) : 0.f;
  bfu* o = out + bn * outRowStride + t;
#pragma unroll
  for (int k = 0; k < K_; ++k) {
    float mask = w[k * 3 + 0] * xm + w[k * 3 + 1] * x0 + w[k * 3 + 2] * xp + b[k];
    o[(long)k * T_] = f2b(x0 * mask);
  }
}

// guided mask: conv over (K,3) neighborhood, mask multiplies x_imf
__global__ void k_guided(const bfu* __restrict__ ximf, const float* __restrict__ w,
                         const float* __restrict__ b, bfu* __restrict__ out)
{
  long idx = (long)blockIdx.x * blockDim.x + threadIdx.x;
  const long total = (long)B_ * N_ * T_;
  if (idx >= total) return;
  long bn = idx / T_; int t = (int)(idx % T_);
  const bfu* xr = ximf + bn * (long)(K_ * T_);
  float v[K_][3];
#pragma unroll
  for (int kk = 0; kk < K_; ++kk) {
    v[kk][0] = t > 0      ? b2f(xr[kk * T_ + t - 1]) : 0.f;
    v[kk][1] =              b2f(xr[kk * T_ + t]);
    v[kk][2] = t < T_ - 1 ? b2f(xr[kk * T_ + t + 1]) : 0.f;
  }
  bfu* o = out + bn * (long)(K_ * T_) + t;
#pragma unroll
  for (int k = 0; k < K_; ++k) {
    float m = b[k];
#pragma unroll
    for (int kk = 0; kk < K_; ++kk)
#pragma unroll
      for (int d = 0; d < 3; ++d)
        m += w[(k * K_ + kk) * 3 + d] * v[kk][d];
    o[(long)k * T_] = f2b(v[k][1] * m);
  }
}

// BatchNorm stats over (outer, C, I) contiguous bf16: one block per channel
__global__ __launch_bounds__(256)
void k_bn_stats(const bfu* __restrict__ x, int C, long I, int outer,
                float* __restrict__ mean, float* __restrict__ inv)
{
  __shared__ float ss[256], sq[256];
  const int c = blockIdx.x;
  float s = 0.f, s2 = 0.f;
  for (int b = 0; b < outer; ++b) {
    const bfu* p = x + ((long)b * C + c) * I;
    for (long i = threadIdx.x; i < I; i += blockDim.x) {
      float v = b2f(p[i]); s += v; s2 += v * v;
    }
  }
  ss[threadIdx.x] = s; sq[threadIdx.x] = s2;
  __syncthreads();
  for (int st = 128; st > 0; st >>= 1) {
    if ((int)threadIdx.x < st) {
      ss[threadIdx.x] += ss[threadIdx.x + st];
      sq[threadIdx.x] += sq[threadIdx.x + st];
    }
    __syncthreads();
  }
  if (threadIdx.x == 0) {
    float n  = (float)outer * (float)I;
    float m  = ss[0] / n;
    float vr = sq[0] / n - m * m;
    mean[c] = m;
    inv[c]  = rsqrtf(vr + EPS_);
  }
}

__global__ void k_bn_leaky(bfu* __restrict__ x, const float* __restrict__ mean,
                           const float* __restrict__ inv, int C, long I, long total)
{
  long idx = (long)blockIdx.x * blockDim.x + threadIdx.x;
  if (idx >= total) return;
  int c = (int)((idx / I) % C);
  x[idx] = f2b(leaky_((b2f(x[idx]) - mean[c]) * inv[c]));
}

// bn+leaky on sel hidden, Linear(64->2), gumbel-softmax (hash uniforms)
__global__ void k_sel_gumbel(const bfu* __restrict__ h, const float* __restrict__ mean,
                             const float* __restrict__ inv, const float* __restrict__ w2,
                             const float* __restrict__ b2, float* __restrict__ hard,
                             unsigned seed)
{
  int idx = blockIdx.x * blockDim.x + threadIdx.x;
  if (idx >= B_ * N_ * K_) return;
  int n = (idx / K_) % N_;
  const bfu* hp = h + (long)idx * 64;
  float m = mean[n], iv = inv[n];
  float a0 = b2[0], a1 = b2[1];
#pragma unroll 8
  for (int c = 0; c < 64; ++c) {
    float v = leaky_((b2f(hp[c]) - m) * iv);
    a0 += w2[c] * v;
    a1 += w2[64 + c] * v;
  }
  unsigned r0 = hash_u32(seed * 0x9E3779B9u + (unsigned)(idx * 2 + 0));
  unsigned r1 = hash_u32(seed * 0x9E3779B9u + (unsigned)(idx * 2 + 1));
  float u0 = (r0 >> 8) * (1.f / 16777216.f) + 1e-10f;
  float u1 = (r1 >> 8) * (1.f / 16777216.f) + 1e-10f;
  float g0 = a0 - logf(-logf(u0));
  float g1 = a1 - logf(-logf(u1));
  float mx = fmaxf(g0, g1);
  float e0 = expf(g0 - mx), e1 = expf(g1 - mx);
  float d  = e0 + e1;
  hard[idx * 2 + 0] = e0 / d;
  hard[idx * 2 + 1] = e1 / d;
}

// x_summed: einsum('bnkt,bnkc->bntc') split into left/right streams
__global__ void k_combine(const bfu* __restrict__ ximf, const float* __restrict__ hard,
                          bfu* __restrict__ outl, bfu* __restrict__ outr)
{
  long idx = (long)blockIdx.x * blockDim.x + threadIdx.x;
  const long total = (long)B_ * N_ * T_;
  if (idx >= total) return;
  long bn = idx / T_; int t = (int)(idx % T_);
  const bfu*   xr = ximf + bn * (long)(K_ * T_);
  const float* hd = hard + bn * (long)(K_ * 2);
  float sl = 0.f, sr = 0.f;
#pragma unroll
  for (int k = 0; k < K_; ++k) {
    float v = b2f(xr[k * T_ + t]);
    sl += v * hd[k * 2 + 0];
    sr += v * hd[k * 2 + 1];
  }
  outl[idx] = f2b(sl);
  outr[idx] = f2b(sr);
}

// h = irfft(S): real symmetric filter from per-frequency scales
__global__ void k_filter(const float* __restrict__ low, const float* __restrict__ high,
                         float* __restrict__ hf)
{
  int d = blockIdx.x * blockDim.x + threadIdx.x;
  if (d >= T_) return;
  float s = 0.f;
  for (int f = 0; f <= 256; ++f) {
    float wf = f < 128 ? low[f] : high[f - 128];
    float cf = (f == 0 || f == 256) ? 1.f : 2.f;
    int fd = (f * d) & 511;   // exact reduction (period 512)
    s += cf * wf * __cosf(6.28318530717958647692f * (float)fd * (1.f / 512.f));
  }
  hf[d] = s * (1.f / 512.f);
}

// Circulant C[t,tau] = h[(t-tau) mod 512] (bf16): fourier_feature == GEMM
__global__ void k_circ(const float* __restrict__ hf, bfu* __restrict__ cm)
{
  int idx = blockIdx.x * blockDim.x + threadIdx.x;
  if (idx >= T_ * T_) return;
  int t = idx >> 9, tau = idx & 511;
  cm[idx] = f2b(hf[(t - tau) & 511]);
}

// ---------------------------------------------------------------------------
// Host orchestration
// ---------------------------------------------------------------------------
struct NodeIdx {
  int mcd_w, mcd_b, bs_w, bs_b, sel_w1, sel_b1, sel_w2, sel_b2,
      rl_w1, rl_b1, rl_w2, rl_b2, rr_w1, rr_b1, rr_w2, rr_b2;
};
struct NodeW { bfu *sel_w1, *rl_w1, *rl_w2, *rr_w1, *rr_w2; };

static void reconstruct_node(void* const* d_in, const NodeIdx& ni, const NodeW& nw,
                             const bfu* xin, bfu* outl, bfu* outr,
                             bfu* ximf, bfu* ximf2, bfu* hsel, float* hard,
                             bfu* hproj, float* mean, float* inv,
                             unsigned seed, hipStream_t stream)
{
  auto F = [&](int i) { return (const float*)d_in[i]; };
  const long BNT = (long)B_ * N_ * T_;
  const int  TPB = 256;
  const int  gBNT = (int)((BNT + TPB - 1) / TPB);

  // mcd -> x_imf
  k_mcd<<<gBNT, TPB, 0, stream>>>(xin, F(ni.mcd_w), F(ni.mcd_b), ximf, (long)K_ * T_);

  // sel Linear(T->64) via WMMA
  dim3 g1((B_ * N_ * K_) / BM, 64 / BN);
  k_gemm<<<g1, 256, 0, stream>>>(ximf, (long)T_, nw.sel_w1, (long)T_,
                                 F(ni.sel_b1), nullptr, 0, hsel, 64, T_, 0, 1);
  k_bn_stats<<<N_, 256, 0, stream>>>(hsel, N_, (long)K_ * 64, B_, mean, inv);
  k_sel_gumbel<<<(B_ * N_ * K_ + TPB - 1) / TPB, TPB, 0, stream>>>(
      hsel, mean, inv, F(ni.sel_w2), F(ni.sel_b2), hard, seed);
  k_guided<<<gBNT, TPB, 0, stream>>>(ximf, F(ni.bs_w), F(ni.bs_b), ximf2);
  k_combine<<<gBNT, TPB, 0, stream>>>(ximf2, hard, outl, outr);

  dim3 g2((B_ * N_) / BM, H_ / BN);
  dim3 g3((B_ * N_) / BM, T_ / BN);
  const long BNH = (long)B_ * N_ * H_;

  // left projection: T->H (bn,leaky) -> T (bn,leaky)
  k_gemm<<<g2, 256, 0, stream>>>(outl, (long)T_, nw.rl_w1, (long)T_,
                                 F(ni.rl_b1), nullptr, 0, hproj, (long)H_, T_, 0, 1);
  k_bn_stats<<<N_, 256, 0, stream>>>(hproj, N_, (long)H_, B_, mean, inv);
  k_bn_leaky<<<(int)((BNH + TPB - 1) / TPB), TPB, 0, stream>>>(hproj, mean, inv, N_, (long)H_, BNH);
  k_gemm<<<g3, 256, 0, stream>>>(hproj, (long)H_, nw.rl_w2, (long)H_,
                                 F(ni.rl_b2), nullptr, 0, outl, (long)T_, H_, 0, 1);
  k_bn_stats<<<N_, 256, 0, stream>>>(outl, N_, (long)T_, B_, mean, inv);
  k_bn_leaky<<<gBNT, TPB, 0, stream>>>(outl, mean, inv, N_, (long)T_, BNT);

  // right projection
  k_gemm<<<g2, 256, 0, stream>>>(outr, (long)T_, nw.rr_w1, (long)T_,
                                 F(ni.rr_b1), nullptr, 0, hproj, (long)H_, T_, 0, 1);
  k_bn_stats<<<N_, 256, 0, stream>>>(hproj, N_, (long)H_, B_, mean, inv);
  k_bn_leaky<<<(int)((BNH + TPB - 1) / TPB), TPB, 0, stream>>>(hproj, mean, inv, N_, (long)H_, BNH);
  k_gemm<<<g3, 256, 0, stream>>>(hproj, (long)H_, nw.rr_w2, (long)H_,
                                 F(ni.rr_b2), nullptr, 0, outr, (long)T_, H_, 0, 1);
  k_bn_stats<<<N_, 256, 0, stream>>>(outr, N_, (long)T_, B_, mean, inv);
  k_bn_leaky<<<gBNT, TPB, 0, stream>>>(outr, mean, inv, N_, (long)T_, BNT);
}

extern "C" void kernel_launch(void* const* d_in, const int* in_sizes, int n_in,
                              void* d_out, int out_size, void* d_ws, size_t ws_size,
                              hipStream_t stream)
{
  (void)in_sizes; (void)n_in; (void)out_size; (void)ws_size;
  auto F = [&](int i) { return (const float*)d_in[i]; };

  float* out = (float*)d_out;

  // bump allocator over d_ws
  char* wp = (char*)d_ws;
  auto alloc = [&](size_t bytes) -> void* {
    void* p = (void*)wp;
    wp += (bytes + 255) & ~(size_t)255;
    return p;
  };
  const long BNT = (long)B_ * N_ * T_;

  bfu* xbf    = (bfu*)alloc((size_t)BNT * sizeof(bfu));
  bfu* ximf   = (bfu*)alloc((size_t)BNT * K_ * sizeof(bfu));
  bfu* ximf2  = (bfu*)alloc((size_t)BNT * K_ * sizeof(bfu));
  bfu* hsel   = (bfu*)alloc((size_t)B_ * N_ * K_ * 64 * sizeof(bfu));
  float* hard = (float*)alloc((size_t)B_ * N_ * K_ * 2 * sizeof(float));
  bfu* hproj  = (bfu*)alloc((size_t)B_ * N_ * H_ * sizeof(bfu));
  float* mean = (float*)alloc((size_t)N_ * sizeof(float));
  float* inv  = (float*)alloc((size_t)N_ * sizeof(float));
  bfu* z2l    = (bfu*)alloc((size_t)BNT * sizeof(bfu));
  bfu* z2r    = (bfu*)alloc((size_t)BNT * sizeof(bfu));
  bfu* z1l    = (bfu*)alloc((size_t)BNT * sizeof(bfu));
  bfu* z1r    = (bfu*)alloc((size_t)BNT * sizeof(bfu));
  bfu* total  = (bfu*)alloc((size_t)BNT * NBR_ * sizeof(bfu));
  bfu* total2 = (bfu*)alloc((size_t)BNT * NBR_ * sizeof(bfu));
  float* hfilt= (float*)alloc((size_t)T_ * sizeof(float));
  bfu* cmat   = (bfu*)alloc((size_t)T_ * T_ * sizeof(bfu));

  // Parameter leaf indices (python-dict insertion order; x is d_in[0]).
  NodeIdx n2  = { 1, 2, 3, 4, 5, 6, 7, 8, 9,10,11,12,13,14,15,16};
  NodeIdx n1l = {17,18,19,20,21,22,23,24,25,26,27,28,29,30,31,32};
  NodeIdx n1r = {37,38,39,40,41,42,43,44,45,46,47,48,49,50,51,52};
  const int llmw = 33, llmb = 34, lrmw = 35, lrmb = 36;   // n1l children mcd
  const int rlmw = 53, rlmb = 54, rrmw = 55, rrmb = 56;   // n1r children mcd
  const int ilow = 57, ihigh = 58, if1w = 59, if1b = 60, if2w = 61, if2b = 62;

  const int TPB = 256;
  const int gBNT = (int)((BNT + TPB - 1) / TPB);

  // one-time fp32 -> bf16 conversions (input + all GEMM weights)
  auto conv = [&](int idx, long n) -> bfu* {
    bfu* d = (bfu*)alloc((size_t)n * sizeof(bfu));
    k_f2b<<<(int)((n + TPB - 1) / TPB), TPB, 0, stream>>>(F(idx), d, n);
    return d;
  };
  k_f2b<<<gBNT, TPB, 0, stream>>>(F(0), xbf, BNT);

  const NodeIdx* nis[3] = {&n2, &n1l, &n1r};
  NodeW nws[3];
  for (int i = 0; i < 3; ++i) {
    const NodeIdx& ni = *nis[i];
    nws[i].sel_w1 = conv(ni.sel_w1, (long)64 * T_);
    nws[i].rl_w1  = conv(ni.rl_w1,  (long)H_ * T_);
    nws[i].rl_w2  = conv(ni.rl_w2,  (long)T_ * H_);
    nws[i].rr_w1  = conv(ni.rr_w1,  (long)H_ * T_);
    nws[i].rr_w2  = conv(ni.rr_w2,  (long)T_ * H_);
  }
  bfu* w_f1 = conv(if1w, (long)NBR_ * H_ * T_);
  bfu* w_f2 = conv(if2w, (long)NBR_ * T_ * H_);

  // level 2
  reconstruct_node(d_in, n2, nws[0], xbf, z2l, z2r, ximf, ximf2, hsel, hard,
                   hproj, mean, inv, 1u, stream);
  // level 1 left; leaves write straight into total slices [0..3], [4..7]
  reconstruct_node(d_in, n1l, nws[1], z2l, z1l, z1r, ximf, ximf2, hsel, hard,
                   hproj, mean, inv, 2u, stream);
  k_mcd<<<gBNT, TPB, 0, stream>>>(z1l, F(llmw), F(llmb), total + 0 * T_, (long)NBR_ * T_);
  k_mcd<<<gBNT, TPB, 0, stream>>>(z1r, F(lrmw), F(lrmb), total + 4 * T_, (long)NBR_ * T_);
  // level 1 right; slices [8..11], [12..15]
  reconstruct_node(d_in, n1r, nws[2], z2r, z1l, z1r, ximf, ximf2, hsel, hard,
                   hproj, mean, inv, 3u, stream);
  k_mcd<<<gBNT, TPB, 0, stream>>>(z1l, F(rlmw), F(rlmb), total + 8 * T_, (long)NBR_ * T_);
  k_mcd<<<gBNT, TPB, 0, stream>>>(z1r, F(rrmw), F(rrmb), total + 12 * T_, (long)NBR_ * T_);

  // fourier feature == circulant matmul; residual add fused in GEMM epilogue
  k_filter<<<2, 256, 0, stream>>>(F(ilow), F(ihigh), hfilt);
  k_circ<<<(T_ * T_) / 256, 256, 0, stream>>>(hfilt, cmat);
  dim3 gf((B_ * N_ * NBR_) / BM, T_ / BN);
  k_gemm<<<gf, 256, 0, stream>>>(total, (long)T_, cmat, (long)T_,
                                 nullptr, total, (long)T_, total2, (long)T_, T_, 0, 1);

  // 16 per-branch heads: T->H leaky, H->T ; strided A/C views into (B,N,16,T)
  dim3 gh1((B_ * N_) / BM, H_ / BN);
  dim3 gh2((B_ * N_) / BM, T_ / BN);
  for (int k = 0; k < NBR_; ++k) {
    k_gemm<<<gh1, 256, 0, stream>>>(total2 + (long)k * T_, (long)NBR_ * T_,
                                    w_f1 + (long)k * H_ * T_, (long)T_,
                                    F(if1b) + (long)k * H_, nullptr, 0,
                                    hproj, (long)H_, T_, 1, 1);
    k_gemm<<<gh2, 256, 0, stream>>>(hproj, (long)H_,
                                    w_f2 + (long)k * T_ * H_, (long)H_,
                                    F(if2b) + (long)k * T_, nullptr, 0,
                                    out + (long)k * T_, (long)NBR_ * T_, H_, 0, 0);
  }
}
